// MaskHeadDeformableConv_12515534701212
// MI455X (gfx1250) — matrix operand order repaired and checked by reading, not compiled, not run on hardware
//
#include <hip/hip_runtime.h>

typedef __attribute__((ext_vector_type(16))) _Float16 v16h;
typedef __attribute__((ext_vector_type(8)))  float    v8f;

#define NINST 8

// ---------------------------------------------------------------------------
// concat: x0[n, 0:256] = compressed (broadcast n), x0[n, 256:264] = bbox_mask
// ---------------------------------------------------------------------------
static __global__ void concat_kernel(const float* __restrict__ comp,
                                     const float* __restrict__ bbox,
                                     float* __restrict__ x0)
{
    const int HW = 24 * 24, C = 264;
    const long long total = (long long)NINST * C * HW;
    for (long long t = blockIdx.x * (long long)blockDim.x + threadIdx.x; t < total;
         t += (long long)gridDim.x * blockDim.x) {
        int j = (int)(t % HW);
        int c = (int)((t / HW) % C);
        int n = (int)(t / ((long long)C * HW));
        x0[t] = (c < 256) ? comp[(long long)c * HW + j]
                          : bbox[((long long)n * 8 + (c - 256)) * HW + j];
    }
}

// ---------------------------------------------------------------------------
// pack main DCN weights (COUT x KTOT fp32 row-major) into WMMA A-fragment
// order (f16): apack[tile][gstep][lane][i], k = gstep*32 + ktmap(lane,i)
// ---------------------------------------------------------------------------
static __global__ void pack_w_kernel(const float* __restrict__ wsrc,
                                     _Float16* __restrict__ apack,
                                     int COUT, int KTOT, int S, int tiles)
{
    long long total = (long long)tiles * S * 32 * 16;
    for (long long t = blockIdx.x * (long long)blockDim.x + threadIdx.x; t < total;
         t += (long long)gridDim.x * blockDim.x) {
        int i    = (int)(t & 15);
        int lane = (int)((t >> 4) & 31);
        int gs   = (int)((t >> 9) % S);
        int tile = (int)(t / ((long long)S * 512));
        int kt = i + (i >= 8 ? 8 : 0) + (lane >= 16 ? 8 : 0);  // 16-bit A 16x32 layout
        int k  = gs * 32 + kt;
        int m  = tile * 16 + (lane & 15);
        float v = (m < COUT && k < KTOT) ? wsrc[(size_t)m * KTOT + k] : 0.0f;
        apack[t] = (_Float16)v;
    }
}

// ---------------------------------------------------------------------------
// pack offset(18) + modulation(9) conv weights into one 32-row (2-tile)
// A-fragment matrix: row m<18 -> ow[m], 18<=m<27 -> mw[m-18], else zero.
// ---------------------------------------------------------------------------
static __global__ void pack_offmod_kernel(const float* __restrict__ ow,
                                          const float* __restrict__ mw,
                                          _Float16* __restrict__ apack,
                                          int KTOT, int S)
{
    long long total = 2LL * S * 512;
    for (long long t = blockIdx.x * (long long)blockDim.x + threadIdx.x; t < total;
         t += (long long)gridDim.x * blockDim.x) {
        int i    = (int)(t & 15);
        int lane = (int)((t >> 4) & 31);
        int gs   = (int)((t >> 9) % S);
        int tile = (int)(t / ((long long)S * 512));
        int kt = i + (i >= 8 ? 8 : 0) + (lane >= 16 ? 8 : 0);
        int k  = gs * 32 + kt;
        int m  = tile * 16 + (lane & 15);
        float v = 0.0f;
        if (k < KTOT) {
            if (m < 18)      v = ow[(size_t)m * KTOT + k];
            else if (m < 27) v = mw[(size_t)(m - 18) * KTOT + k];
        }
        apack[t] = (_Float16)v;
    }
}

// ---------------------------------------------------------------------------
// offset+modulation 3x3 conv as WMMA GEMM (27 outputs padded to 32 = 2 tiles).
// B tile is a zero-padded im2col read staged to LDS. Epilogue adds bias and
// applies 2*sigmoid for the modulation rows, writing OFF/MOD layouts.
// grid: (ceil(HW/16), NINST), block: 64
// ---------------------------------------------------------------------------
template <int CIN, int H, int W>
static __global__ void offmod_gemm_kernel(const float* __restrict__ x,
                                          const _Float16* __restrict__ apack,
                                          const float* __restrict__ ob,
                                          const float* __restrict__ mb,
                                          float* __restrict__ off,
                                          float* __restrict__ mod)
{
    const int HW   = H * W;
    const int CH   = (CIN + 31) / 32;
    const int S    = CH * 9;
    const int nthr = 64;

    const int n    = blockIdx.y;
    const int j0   = blockIdx.x * 16;
    const int tid  = threadIdx.x;
    const int lane = tid & 31;
    const int wave = tid >> 5;

    __shared__ int   sIdx[9][16];
    __shared__ float sWt [9][16];
    __shared__ alignas(32) _Float16 Bs[288 * 16];

    for (int t = tid; t < 9 * 16; t += nthr) {
        int kk = t >> 4, p = t & 15;
        int j = j0 + p;
        int id = 0; float wv = 0.f;
        if (j < HW) {
            int h = j / W, w = j % W;
            int yy = h + kk / 3 - 1, xx = w + kk % 3 - 1;
            if (yy >= 0 && yy < H && xx >= 0 && xx < W) { id = yy * W + xx; wv = 1.f; }
        }
        sIdx[kk][p] = id; sWt[kk][p] = wv;
    }
    __syncthreads();

    v8f acc = {0.f, 0.f, 0.f, 0.f, 0.f, 0.f, 0.f, 0.f};

    for (int q = 0; q < CH; ++q) {
        for (int t = tid; t < 288 * 16; t += nthr) {
            int r = t >> 4, p = t & 15;
            int k = q * 288 + r;
            int c = k / 9, kk = k % 9;
            float v = 0.f;
            if (c < CIN)
                v = sWt[kk][p] * x[((size_t)n * CIN + c) * HW + sIdx[kk][p]];
            Bs[((r >> 5) * 32 + p + (((r >> 3) & 1) << 4)) * 16 + (r & 7) + (((r >> 4) & 1) << 3)] = (_Float16)v;
        }
        __syncthreads();
#pragma unroll
        for (int s = 0; s < 9; ++s) {
            int gs = q * 9 + s;
            v16h a = *(const v16h*)(apack + (((size_t)wave * S + gs) * 32 + lane) * 16);
            v16h b = *(const v16h*)(&Bs[(s * 32 + lane) * 16]);
            acc = __builtin_amdgcn_wmma_f32_16x16x32_f16(false, a, false, b,
                                                         (short)0, acc, false, false);
        }
        __syncthreads();
    }

    int j = j0 + (lane & 15);
    int mbase = wave * 16 + ((lane >> 4) << 3);
    if (j < HW) {
#pragma unroll
        for (int r = 0; r < 8; ++r) {
            int m = mbase + r;
            if (m < 18) {
                off[((size_t)n * 18 + m) * HW + j] = acc[r] + ob[m];
            } else if (m < 27) {
                float a2 = acc[r] + mb[m - 18];
                mod[((size_t)n * 9 + (m - 18)) * HW + j] = 2.0f / (1.0f + expf(-a2));
            }
        }
    }
}

// ---------------------------------------------------------------------------
// Fused deformable-conv GEMM: D(16 couts x 16 pixels) per wave, bilinear
// gather staged to LDS in 288-row K chunks. grid: (ceil(HW/16), NINST),
// block: 32 * ceil(COUT/16) threads.
// ---------------------------------------------------------------------------
template <int CIN, int COUT, int H, int W>
static __global__ void dcn_gemm_kernel(const float* __restrict__ x,
                                       const float* __restrict__ offp,
                                       const float* __restrict__ modp,
                                       const _Float16* __restrict__ apack,
                                       float* __restrict__ y)
{
    const int HW   = H * W;
    const int CH   = (CIN + 31) / 32;   // K chunks (32 channels x 9 taps each)
    const int S    = CH * 9;            // total 32-wide K steps
    const int NW   = (COUT + 15) / 16;  // waves per block
    const int nthr = 32 * NW;

    const int n    = blockIdx.y;
    const int j0   = blockIdx.x * 16;
    const int tid  = threadIdx.x;
    const int lane = tid & 31;
    const int wave = tid >> 5;

    __shared__ int   sIdx[9][16][4];
    __shared__ float sWt [9][16][4];
    __shared__ alignas(32) _Float16 Bs[288 * 16];

    // --- per-block geometry: bilinear corner indices + (weight * valid * mod)
    for (int t = tid; t < 9 * 16; t += nthr) {
        int kk = t >> 4, p = t & 15;
        int j = j0 + p;
        int   id0 = 0, id1 = 0, id2 = 0, id3 = 0;
        float w0 = 0.f, w1 = 0.f, w2 = 0.f, w3 = 0.f;
        if (j < HW) {
            int h = j / W, w = j % W;
            float dy = offp[((size_t)n * 18 + 2 * kk)     * HW + j];
            float dx = offp[((size_t)n * 18 + 2 * kk + 1) * HW + j];
            float m  = modp[((size_t)n * 9 + kk) * HW + j];
            float ys = (float)(h - 1 + kk / 3) + dy;
            float xs = (float)(w - 1 + kk % 3) + dx;
            float y0f = floorf(ys), x0f = floorf(xs);
            float wy = ys - y0f, wx = xs - x0f;
            auto corner = [&](float yc, float xc, float wgt, int& id, float& wo) {
                bool valid = (yc >= 0.f) && (yc <= (float)(H - 1)) &&
                             (xc >= 0.f) && (xc <= (float)(W - 1));
                int yi = (int)fminf(fmaxf(yc, 0.f), (float)(H - 1));
                int xi = (int)fminf(fmaxf(xc, 0.f), (float)(W - 1));
                id = yi * W + xi;
                wo = valid ? wgt * m : 0.f;
            };
            corner(y0f,       x0f,       (1.f - wy) * (1.f - wx), id0, w0);
            corner(y0f,       x0f + 1.f, (1.f - wy) * wx,         id1, w1);
            corner(y0f + 1.f, x0f,       wy * (1.f - wx),         id2, w2);
            corner(y0f + 1.f, x0f + 1.f, wy * wx,                 id3, w3);
        }
        sIdx[kk][p][0] = id0; sIdx[kk][p][1] = id1; sIdx[kk][p][2] = id2; sIdx[kk][p][3] = id3;
        sWt [kk][p][0] = w0;  sWt [kk][p][1] = w1;  sWt [kk][p][2] = w2;  sWt [kk][p][3] = w3;
    }
    __syncthreads();

    v8f acc = {0.f, 0.f, 0.f, 0.f, 0.f, 0.f, 0.f, 0.f};

    for (int q = 0; q < CH; ++q) {
        // cooperative gather of B chunk (288 x 16, f16) into LDS
        for (int t = tid; t < 288 * 16; t += nthr) {
            int r = t >> 4, p = t & 15;
            int k = q * 288 + r;           // flat K index (== c*9+kk)
            int c = k / 9, kk = k % 9;
            float v = 0.f;
            if (c < CIN) {
                const float* xb = x + ((size_t)n * CIN + c) * HW;
                v = sWt[kk][p][0] * xb[sIdx[kk][p][0]]
                  + sWt[kk][p][1] * xb[sIdx[kk][p][1]]
                  + sWt[kk][p][2] * xb[sIdx[kk][p][2]]
                  + sWt[kk][p][3] * xb[sIdx[kk][p][3]];
            }
            Bs[((r >> 5) * 32 + p + (((r >> 3) & 1) << 4)) * 16 + (r & 7) + (((r >> 4) & 1) << 3)] = (_Float16)v;
        }
        __syncthreads();
#pragma unroll
        for (int s = 0; s < 9; ++s) {
            int gs = q * 9 + s;
            v16h a = *(const v16h*)(apack + (((size_t)wave * S + gs) * 32 + lane) * 16);
            v16h b = *(const v16h*)(&Bs[(s * 32 + lane) * 16]);
            acc = __builtin_amdgcn_wmma_f32_16x16x32_f16(false, a, false, b,
                                                         (short)0, acc, false, false);
        }
        __syncthreads();
    }

    // store D: VGPR r -> M = wave*16 + r + (lane>=16 ? 8 : 0), N = lane&15
    int j = j0 + (lane & 15);
    int mbase = wave * 16 + ((lane >> 4) << 3);
    if (j < HW) {
#pragma unroll
        for (int r = 0; r < 8; ++r) {
            int m = mbase + r;
            if (m < COUT) y[((size_t)n * COUT + m) * HW + j] = acc[r];
        }
    }
}

// ---------------------------------------------------------------------------
// 1x1 conv (FPN adapter) as WMMA GEMM: out[COUT x HW] = w[COUT x CIN] * fpn + b
// grid: HW/16 blocks, block: 32 * (COUT/16) threads. N=1 batch.
// ---------------------------------------------------------------------------
template <int CIN, int COUT, int HW>
static __global__ void conv1x1_gemm_kernel(const float* __restrict__ fpn,
                                           const _Float16* __restrict__ apack,
                                           const float* __restrict__ bias,
                                           float* __restrict__ out)
{
    const int S    = CIN / 32;
    const int lane = threadIdx.x & 31;
    const int wave = threadIdx.x >> 5;
    const int p0   = blockIdx.x * 16;

    v8f acc = {0.f, 0.f, 0.f, 0.f, 0.f, 0.f, 0.f, 0.f};
    for (int s = 0; s < S; ++s) {
        v16h a = *(const v16h*)(apack + (((size_t)wave * S + s) * 32 + lane) * 16);
        const float* bp = fpn + (size_t)(s * 32) * HW + p0 + (lane & 15);
        v16h b;
#pragma unroll
        for (int i = 0; i < 16; ++i) b[i] = (_Float16)bp[(size_t)(i + (i >= 8 ? 8 : 0) + ((lane >> 4) << 3)) * HW];
        acc = __builtin_amdgcn_wmma_f32_16x16x32_f16(false, a, false, b,
                                                     (short)0, acc, false, false);
    }
    int p = p0 + (lane & 15);
    int mbase = wave * 16 + ((lane >> 4) << 3);
#pragma unroll
    for (int r = 0; r < 8; ++r) {
        int m = mbase + r;
        out[(size_t)m * HW + p] = acc[r] + bias[m];
    }
}

// ---------------------------------------------------------------------------
// GroupNorm (8 groups, biased var) + ReLU (+ optional broadcast extra add)
// grid: NINST*8 blocks, block: 256
// ---------------------------------------------------------------------------
template <bool ADD_EXTRA>
static __global__ void gn_kernel(const float* __restrict__ y, float* __restrict__ xout,
                                 const float* __restrict__ g, const float* __restrict__ b,
                                 const float* __restrict__ extra, int C, int HW)
{
    const int G = 8;
    int n  = blockIdx.x / G;
    int gr = blockIdx.x % G;
    int Cg = C / G;
    size_t base = ((size_t)n * C + (size_t)gr * Cg) * HW;
    int cnt = Cg * HW;

    float s = 0.f, s2 = 0.f;
    for (int i = threadIdx.x; i < cnt; i += blockDim.x) {
        float v = y[base + i];
        s += v; s2 += v * v;
    }
    __shared__ float rs_[256], rq_[256];
    rs_[threadIdx.x] = s; rq_[threadIdx.x] = s2;
    __syncthreads();
    for (int off = 128; off > 0; off >>= 1) {
        if ((int)threadIdx.x < off) {
            rs_[threadIdx.x] += rs_[threadIdx.x + off];
            rq_[threadIdx.x] += rq_[threadIdx.x + off];
        }
        __syncthreads();
    }
    float mu   = rs_[0] / (float)cnt;
    float var  = rq_[0] / (float)cnt - mu * mu;
    float rstd = rsqrtf(var + 1e-5f);
    for (int i = threadIdx.x; i < cnt; i += blockDim.x) {
        int c = gr * Cg + i / HW;
        float v = (y[base + i] - mu) * rstd * g[c] + b[c];
        v = fmaxf(v, 0.f);
        if (ADD_EXTRA) v += extra[(size_t)c * HW + (i % HW)];
        xout[base + i] = v;
    }
}

// ---------------------------------------------------------------------------
// xnew[n,c,h2,w2] = adapter[c,h2,w2] (broadcast n) + xold[n,c,h2/2,w2/2]
// ---------------------------------------------------------------------------
static __global__ void up2add_kernel(const float* __restrict__ ad,
                                     const float* __restrict__ xold,
                                     float* __restrict__ xnew,
                                     int C, int H2, int W2)
{
    int HW2 = H2 * W2;
    long long total = (long long)NINST * C * HW2;
    int H1 = H2 >> 1, W1 = W2 >> 1;
    for (long long t = blockIdx.x * (long long)blockDim.x + threadIdx.x; t < total;
         t += (long long)gridDim.x * blockDim.x) {
        int j = (int)(t % HW2);
        int c = (int)((t / HW2) % C);
        int n = (int)(t / ((long long)C * HW2));
        int h2 = j / W2, w2 = j % W2;
        xnew[t] = ad[(size_t)c * HW2 + j] +
                  xold[(((size_t)n * C + c) * H1 + (h2 >> 1)) * W1 + (w2 >> 1)];
    }
}

// ---------------------------------------------------------------------------
// final 3x3 conv 16 -> 1, pad 1. grid: (HW/256, NINST), block 256
// ---------------------------------------------------------------------------
static __global__ void outconv_kernel(const float* __restrict__ x,
                                      const float* __restrict__ w,
                                      const float* __restrict__ b,
                                      float* __restrict__ out)
{
    const int H = 192, W = 192, HW = H * W, CI = 16;
    int j = blockIdx.x * blockDim.x + threadIdx.x;
    int n = blockIdx.y;
    if (j >= HW) return;
    int h = j / W, ww = j % W;
    float acc = b[0];
    for (int c = 0; c < CI; ++c) {
        const float* xc = x + ((size_t)n * CI + c) * HW;
        const float* wc = w + c * 9;
#pragma unroll
        for (int dy = 0; dy < 3; ++dy) {
            int yy = h + dy - 1;
            if (yy < 0 || yy >= H) continue;
#pragma unroll
            for (int dx = 0; dx < 3; ++dx) {
                int xx = ww + dx - 1;
                if (xx < 0 || xx >= W) continue;
                acc = fmaf(xc[yy * W + xx], wc[dy * 3 + dx], acc);
            }
        }
    }
    out[(size_t)n * HW + j] = acc;
}

// ---------------------------------------------------------------------------
// host orchestration
// ---------------------------------------------------------------------------
extern "C" void kernel_launch(void* const* d_in, const int* in_sizes, int n_in,
                              void* d_out, int out_size, void* d_ws, size_t ws_size,
                              hipStream_t stream)
{
    (void)in_sizes; (void)n_in; (void)out_size; (void)ws_size;

    const float* compressed = (const float*)d_in[0];
    const float* bbox       = (const float*)d_in[1];
    const float* fpn1       = (const float*)d_in[2];
    const float* fpn2       = (const float*)d_in[3];
    const float* fpn3       = (const float*)d_in[4];
    const float* extra      = (const float*)d_in[5];

    struct DcnP { const float *ow, *ob, *mw, *mb, *w; };
    DcnP L1 = {(const float*)d_in[6],  (const float*)d_in[7],  (const float*)d_in[8],
               (const float*)d_in[9],  (const float*)d_in[10]};
    const float *g1 = (const float*)d_in[11], *b1 = (const float*)d_in[12];
    DcnP L2 = {(const float*)d_in[13], (const float*)d_in[14], (const float*)d_in[15],
               (const float*)d_in[16], (const float*)d_in[17]};
    const float *g2 = (const float*)d_in[18], *b2 = (const float*)d_in[19];
    const float *ad1w = (const float*)d_in[20], *ad1b = (const float*)d_in[21];
    DcnP L3 = {(const float*)d_in[22], (const float*)d_in[23], (const float*)d_in[24],
               (const float*)d_in[25], (const float*)d_in[26]};
    const float *g3 = (const float*)d_in[27], *b3 = (const float*)d_in[28];
    const float *ad2w = (const float*)d_in[29], *ad2b = (const float*)d_in[30];
    DcnP L4 = {(const float*)d_in[31], (const float*)d_in[32], (const float*)d_in[33],
               (const float*)d_in[34], (const float*)d_in[35]};
    const float *g4 = (const float*)d_in[36], *b4 = (const float*)d_in[37];
    const float *ad3w = (const float*)d_in[38], *ad3b = (const float*)d_in[39];
    DcnP L5 = {(const float*)d_in[40], (const float*)d_in[41], (const float*)d_in[42],
               (const float*)d_in[43], (const float*)d_in[44]};
    const float *g5 = (const float*)d_in[45], *b5 = (const float*)d_in[46];
    const float *outw = (const float*)d_in[47], *outb = (const float*)d_in[48];

    float* ws = (float*)d_ws;
    float* XA  = ws;                 // max 4.72M floats
    float* XB  = ws + 5000000;       // max 9.44M floats
    float* Y   = ws + 14600000;      // max 4.72M floats
    float* OFF = ws + 19600000;      // max 5.31M floats
    float* MOD = ws + 25000000;      // max 2.66M floats
    float* AD  = ws + 27700000;      // max 1.18M floats
    _Float16* AP  = (_Float16*)(ws + 28900000);  // main packed weights (~0.71M halfs)
    _Float16* AP2 = (_Float16*)(ws + 29300000);  // offmod packed weights (~83K halfs)

    // ---- stage 0: concat
    concat_kernel<<<1024, 256, 0, stream>>>(compressed, bbox, XA);

    // ---- layer 1: DCN 264 -> 264 @ 24x24
    pack_offmod_kernel<<<(2 * 81 * 512 + 255) / 256, 256, 0, stream>>>(L1.ow, L1.mw, AP2, 2376, 81);
    pack_w_kernel<<<(17 * 81 * 512 + 255) / 256, 256, 0, stream>>>(L1.w, AP, 264, 2376, 81, 17);
    offmod_gemm_kernel<264, 24, 24><<<dim3(36, NINST), 64, 0, stream>>>(
        XA, AP2, L1.ob, L1.mb, OFF, MOD);
    dcn_gemm_kernel<264, 264, 24, 24><<<dim3(36, NINST), 17 * 32, 0, stream>>>(
        XA, OFF, MOD, AP, Y);
    gn_kernel<true><<<NINST * 8, 256, 0, stream>>>(Y, XB, g1, b1, extra, 264, 576);

    // ---- layer 2: DCN 264 -> 128 @ 24x24
    pack_offmod_kernel<<<(2 * 81 * 512 + 255) / 256, 256, 0, stream>>>(L2.ow, L2.mw, AP2, 2376, 81);
    pack_w_kernel<<<(8 * 81 * 512 + 255) / 256, 256, 0, stream>>>(L2.w, AP, 128, 2376, 81, 8);
    offmod_gemm_kernel<264, 24, 24><<<dim3(36, NINST), 64, 0, stream>>>(
        XB, AP2, L2.ob, L2.mb, OFF, MOD);
    dcn_gemm_kernel<264, 128, 24, 24><<<dim3(36, NINST), 8 * 32, 0, stream>>>(
        XB, OFF, MOD, AP, Y);
    gn_kernel<false><<<NINST * 8, 256, 0, stream>>>(Y, XA, g2, b2, nullptr, 128, 576);

    // ---- adapter 1 (1024 -> 128 @ 48x48) + up2 add
    pack_w_kernel<<<(8 * 32 * 512 + 255) / 256, 256, 0, stream>>>(ad1w, AP, 128, 1024, 32, 8);
    conv1x1_gemm_kernel<1024, 128, 2304><<<144, 8 * 32, 0, stream>>>(fpn1, AP, ad1b, AD);
    up2add_kernel<<<2048, 256, 0, stream>>>(AD, XA, XB, 128, 48, 48);

    // ---- layer 3: DCN 128 -> 64 @ 48x48
    pack_offmod_kernel<<<(2 * 36 * 512 + 255) / 256, 256, 0, stream>>>(L3.ow, L3.mw, AP2, 1152, 36);
    pack_w_kernel<<<(4 * 36 * 512 + 255) / 256, 256, 0, stream>>>(L3.w, AP, 64, 1152, 36, 4);
    offmod_gemm_kernel<128, 48, 48><<<dim3(144, NINST), 64, 0, stream>>>(
        XB, AP2, L3.ob, L3.mb, OFF, MOD);
    dcn_gemm_kernel<128, 64, 48, 48><<<dim3(144, NINST), 4 * 32, 0, stream>>>(
        XB, OFF, MOD, AP, Y);
    gn_kernel<false><<<NINST * 8, 256, 0, stream>>>(Y, XA, g3, b3, nullptr, 64, 2304);

    // ---- adapter 2 (512 -> 64 @ 96x96) + up2 add
    pack_w_kernel<<<(4 * 16 * 512 + 255) / 256, 256, 0, stream>>>(ad2w, AP, 64, 512, 16, 4);
    conv1x1_gemm_kernel<512, 64, 9216><<<576, 4 * 32, 0, stream>>>(fpn2, AP, ad2b, AD);
    up2add_kernel<<<4096, 256, 0, stream>>>(AD, XA, XB, 64, 96, 96);

    // ---- layer 4: DCN 64 -> 32 @ 96x96
    pack_offmod_kernel<<<(2 * 18 * 512 + 255) / 256, 256, 0, stream>>>(L4.ow, L4.mw, AP2, 576, 18);
    pack_w_kernel<<<(2 * 18 * 512 + 255) / 256, 256, 0, stream>>>(L4.w, AP, 32, 576, 18, 2);
    offmod_gemm_kernel<64, 96, 96><<<dim3(576, NINST), 64, 0, stream>>>(
        XB, AP2, L4.ob, L4.mb, OFF, MOD);
    dcn_gemm_kernel<64, 32, 96, 96><<<dim3(576, NINST), 2 * 32, 0, stream>>>(
        XB, OFF, MOD, AP, Y);
    gn_kernel<false><<<NINST * 8, 256, 0, stream>>>(Y, XA, g4, b4, nullptr, 32, 9216);

    // ---- adapter 3 (256 -> 32 @ 192x192) + up2 add
    pack_w_kernel<<<(2 * 8 * 512 + 255) / 256, 256, 0, stream>>>(ad3w, AP, 32, 256, 8, 2);
    conv1x1_gemm_kernel<256, 32, 36864><<<2304, 2 * 32, 0, stream>>>(fpn3, AP, ad3b, AD);
    up2add_kernel<<<8192, 256, 0, stream>>>(AD, XA, XB, 32, 192, 192);

    // ---- layer 5: DCN 32 -> 16 @ 192x192
    pack_offmod_kernel<<<(2 * 9 * 512 + 255) / 256, 256, 0, stream>>>(L5.ow, L5.mw, AP2, 288, 9);
    pack_w_kernel<<<(1 * 9 * 512 + 255) / 256, 256, 0, stream>>>(L5.w, AP, 16, 288, 9, 1);
    offmod_gemm_kernel<32, 192, 192><<<dim3(2304, NINST), 64, 0, stream>>>(
        XB, AP2, L5.ob, L5.mb, OFF, MOD);
    dcn_gemm_kernel<32, 16, 192, 192><<<dim3(2304, NINST), 1 * 32, 0, stream>>>(
        XB, OFF, MOD, AP, Y);
    gn_kernel<false><<<NINST * 8, 256, 0, stream>>>(Y, XA, g5, b5, nullptr, 16, 36864);

    // ---- final 3x3 conv -> d_out (8,1,192,192)
    outconv_kernel<<<dim3(144, NINST), 256, 0, stream>>>(XA, outw, outb, (float*)d_out);
}
